// LSTM_Decoder_new_x_80753975099927
// MI455X (gfx1250) — compile-verified, matrix operand order checked by reading, not run on hardware
//
#include <hip/hip_runtime.h>
#include <math.h>

// Pointer-generator LSTM decoder step for MI455X (gfx1250, wave32).
// All dense GEMMs use v_wmma_f32_16x16x32_f16 (fp32->f16 on the fly, fp32 acc).
// Attention score GEMMs are fused with the tanh + v-dot reduction so the
// [B*TK,H] intermediate never hits HBM (saves ~100MB of the ~300MB total).

#define DEV __device__ __forceinline__

typedef __attribute__((ext_vector_type(16))) _Float16 v16h;
typedef __attribute__((ext_vector_type(8)))  float    v8f;

namespace {
constexpr int B_ = 128;
constexpr int TK = 400;
constexpr int NK = 50;
constexpr int H  = 256;
constexpr int E  = 256;
constexpr int V  = 50000;
}

DEV float sigmoidf(float x) { return 1.0f / (1.0f + expf(-x)); }

// ---- WMMA fragment helpers (ISA 7.12.2 layouts, wave32) -------------------
// A (16x32 f16): lane L holds row m0+(L&15); elems 0..7 -> K = kh+0..7,
// elems 8..15 -> K = kh+16..23, where kh = (L>=16 ? 8 : 0).
DEV v16h load_frag_a(const float* __restrict__ A, int lda, int m0, int k0, int lane) {
  const float* p = A + (size_t)(m0 + (lane & 15)) * lda + k0 + ((lane >> 4) << 3);
  v16h f;
#pragma unroll
  for (int j = 0; j < 8; ++j) f[j] = (_Float16)p[j];
#pragma unroll
  for (int j = 0; j < 8; ++j) f[j + 8] = (_Float16)p[j + 16];
  return f;
}

// B (32x16 f16), row-major [K,N] in memory: lane L holds column n0+(L&15),
// K pattern mirrors A.
DEV v16h load_frag_b(const float* __restrict__ Bm, int ldb, int k0, int n0, int lane) {
  const float* p = Bm + (size_t)(k0 + ((lane >> 4) << 3)) * ldb + n0 + (lane & 15);
  v16h f;
#pragma unroll
  for (int j = 0; j < 8; ++j) f[j] = (_Float16)p[(size_t)j * ldb];
#pragma unroll
  for (int j = 0; j < 8; ++j) f[j + 8] = (_Float16)p[(size_t)(j + 16) * ldb];
  return f;
}

DEV v8f wmma32(v16h a, v16h b, v8f c) {
  return __builtin_amdgcn_wmma_f32_16x16x32_f16(false, a, false, b, (short)0, c,
                                                false, false);
}

// ---- Generic WMMA GEMM: C = A@B (+ A2@B2) + bias (+ bias2) ----------------
// Wave-per-16x16 output tile. C/D layout: VGPR r, lane L -> row m0+r+8*(L>>4),
// col n0+(L&15).
__global__ void k_wmma_gemm(const float* __restrict__ A, int lda,
                            const float* __restrict__ Bm, int ldb,
                            const float* __restrict__ A2, int lda2,
                            const float* __restrict__ B2, int ldb2,
                            const float* __restrict__ bias,
                            const float* __restrict__ bias2,
                            float* __restrict__ C, int ldc,
                            int M, int N, int K, int K2) {
  const int lane = threadIdx.x & 31;
  const int wave = threadIdx.x >> 5;
  const int tilesN = N >> 4;
  const int tile = blockIdx.x * (blockDim.x >> 5) + wave;
  if (tile >= (M >> 4) * tilesN) return;
  const int m0 = (tile / tilesN) << 4;
  const int n0 = (tile % tilesN) << 4;
  v8f acc = {};
  for (int k0 = 0; k0 < K; k0 += 32) {
    v16h a = load_frag_a(A, lda, m0, k0, lane);
    v16h b = load_frag_b(Bm, ldb, k0, n0, lane);
    acc = wmma32(a, b, acc);
  }
  if (A2) {
    for (int k0 = 0; k0 < K2; k0 += 32) {
      v16h a = load_frag_a(A2, lda2, m0, k0, lane);
      v16h b = load_frag_b(B2, ldb2, k0, n0, lane);
      acc = wmma32(a, b, acc);
    }
  }
  const int n = n0 + (lane & 15);
  float bv = 0.f;
  if (bias)  bv += bias[n];
  if (bias2) bv += bias2[n];
  const int rb = m0 + ((lane >> 4) << 3);
#pragma unroll
  for (int r = 0; r < 8; ++r) C[(size_t)(rb + r) * ldc + n] = acc[r] + bv;
}

// ---- Fused attention score: e[row] = v . tanh(X@W + sW[b] + cov*wc + bvec) -
// X: [R,H] (rows = b*P + t), W: [H,H]. Wave computes 16 rows: 16 N-tiles x
// 8 K-steps of WMMA, epilogue reduces across the 16 lanes of each half.
__global__ void k_attn_score(const float* __restrict__ X,
                             const float* __restrict__ W,
                             const float* __restrict__ sW,
                             const float* __restrict__ cov,
                             const float* __restrict__ wc,
                             const float* __restrict__ bvec,
                             const float* __restrict__ vred,
                             float* __restrict__ eout,
                             int R, int P) {
  const int lane = threadIdx.x & 31;
  const int wave = threadIdx.x >> 5;
  const int tile = blockIdx.x * (blockDim.x >> 5) + wave;
  const int m0 = tile << 4;
  if (m0 >= R) return;

  v16h afr[8];
#pragma unroll
  for (int ks = 0; ks < 8; ++ks) afr[ks] = load_frag_a(X, H, m0, ks * 32, lane);

  const int half = lane >> 4;
  int gmr[8], br[8];
  float covr[8];
#pragma unroll
  for (int r = 0; r < 8; ++r) {
    gmr[r] = m0 + r + half * 8;
    br[r] = gmr[r] / P;
    covr[r] = cov ? cov[gmr[r]] : 0.f;
  }
  float ep[8] = {};
  for (int nt = 0; nt < 16; ++nt) {
    const int n0 = nt << 4;
    v8f acc = {};
#pragma unroll
    for (int ks = 0; ks < 8; ++ks) {
      v16h bfr = load_frag_b(W, H, ks * 32, n0, lane);
      acc = wmma32(afr[ks], bfr, acc);
    }
    const int n = n0 + (lane & 15);
    const float wcn = wc ? wc[n] : 0.f;
    const float bn = bvec[n];
    const float vn = vred[n];
#pragma unroll
    for (int r = 0; r < 8; ++r) {
      float val = acc[r] + sW[br[r] * H + n] + covr[r] * wcn + bn;
      ep[r] += vn * tanhf(val);
    }
  }
#pragma unroll
  for (int r = 0; r < 8; ++r) {
#pragma unroll
    for (int m = 1; m < 16; m <<= 1) ep[r] += __shfl_xor(ep[r], m, 32);
  }
  if ((lane & 15) == 0) {
#pragma unroll
    for (int r = 0; r < 8; ++r) eout[gmr[r]] = ep[r];
  }
}

// ---- Small elementwise / per-batch kernels --------------------------------
__global__ void k_build_xcat(const float* __restrict__ cd, const float* __restrict__ cg,
                             const float* __restrict__ cg2, const int* __restrict__ y,
                             const float* __restrict__ emb, float* __restrict__ xcat) {
  int b = blockIdx.x, h = threadIdx.x;
  float* row = xcat + (size_t)b * 4 * H;
  row[h]         = cd[b * H + h];
  row[H + h]     = cg[b * H + h];
  row[2 * H + h] = cg2[b * H + h];
  row[3 * H + h] = emb[(size_t)y[b] * E + h];
}

__global__ void k_lstm(const float* __restrict__ gates, const float* __restrict__ c_prev,
                       float* __restrict__ h_out, float* __restrict__ c_out,
                       float* __restrict__ s_hat) {
  int b = blockIdx.x, h = threadIdx.x;
  const float* g = gates + (size_t)b * 4 * H;
  float gi = g[h], gf = g[H + h], gg = g[2 * H + h], go = g[3 * H + h];
  float c = sigmoidf(gf) * c_prev[b * H + h] + sigmoidf(gi) * tanhf(gg);
  float hn = sigmoidf(go) * tanhf(c);
  h_out[b * H + h] = hn;
  c_out[b * H + h] = c;
  s_hat[b * 2 * H + h] = hn;
  s_hat[b * 2 * H + H + h] = c;
}

__global__ void k_doc_attn(const float* __restrict__ e, const float* __restrict__ mask,
                           const float* __restrict__ cov, const float* __restrict__ enc,
                           float* __restrict__ a_out, float* __restrict__ cov_out,
                           float* __restrict__ cd_out) {
  __shared__ float sa[TK];
  __shared__ float sc;
  int b = blockIdx.x, tid = threadIdx.x;
  for (int t = tid; t < TK; t += blockDim.x) sa[t] = e[b * TK + t];
  __syncthreads();
  if (tid == 0) {
    float m = -1e30f;
    for (int t = 0; t < TK; ++t) m = fmaxf(m, sa[t]);
    sc = m;
  }
  __syncthreads();
  float m = sc;
  for (int t = tid; t < TK; t += blockDim.x)
    sa[t] = expf(sa[t] - m) * mask[b * TK + t];
  __syncthreads();
  if (tid == 0) {
    float s = 0.f;
    for (int t = 0; t < TK; ++t) s += sa[t];
    sc = 1.0f / s;
  }
  __syncthreads();
  float inv = sc;
  for (int t = tid; t < TK; t += blockDim.x) {
    float av = sa[t] * inv;
    sa[t] = av;
    a_out[b * TK + t] = av;
    cov_out[b * TK + t] = cov[b * TK + t] + av;
  }
  __syncthreads();
  int h = tid;  // 256 threads
  float acc = 0.f;
  for (int t = 0; t < TK; ++t)
    acc += sa[t] * enc[((size_t)b * TK + t) * H + h];
  cd_out[b * H + h] = acc;
}

__global__ void k_node_attn(const float* __restrict__ e, const float* __restrict__ maskn,
                            const float* __restrict__ flow, const float* __restrict__ encn,
                            const float* __restrict__ graph, const float* __restrict__ n2t,
                            const float* __restrict__ a_doc,
                            float* __restrict__ cg_out, float* __restrict__ cg2_out,
                            float* __restrict__ an2t_out, float* __restrict__ flow_out) {
  __shared__ float san[NK];
  __shared__ float san2[NK];
  __shared__ float sat[TK];
  __shared__ float sc;
  int b = blockIdx.x, tid = threadIdx.x;
  if (tid < NK) san[tid] = e[b * NK + tid];
  __syncthreads();
  if (tid == 0) {
    float m = -1e30f;
    for (int n = 0; n < NK; ++n) m = fmaxf(m, san[n]);
    sc = m;
  }
  __syncthreads();
  if (tid < NK) san[tid] = expf(san[tid] - sc) * maskn[b * NK + tid];
  __syncthreads();
  if (tid == 0) {
    float s = 0.f;
    for (int n = 0; n < NK; ++n) s += san[n];
    sc = 1.0f / s;
  }
  __syncthreads();
  if (tid < NK) san[tid] = san[tid] * sc * flow[b * NK + tid];
  __syncthreads();
  if (tid == 0) {
    float s = 0.f;
    for (int n = 0; n < NK; ++n) s += san[n];
    sc = 1.0f / s;
  }
  __syncthreads();
  if (tid < NK) san[tid] *= sc;
  __syncthreads();
  {  // c_g
    float acc = 0.f;
    for (int n = 0; n < NK; ++n)
      acc += san[n] * encn[((size_t)b * NK + n) * H + tid];
    cg_out[b * H + tid] = acc;
  }
  if (tid < NK) {  // one-hop: an2 = an @ graph, masked + renorm
    float acc = 0.f;
    for (int n = 0; n < NK; ++n)
      acc += san[n] * graph[((size_t)b * NK + n) * NK + tid];
    san2[tid] = acc * maskn[b * NK + tid];
  }
  __syncthreads();
  if (tid == 0) {
    float s = 0.f;
    for (int n = 0; n < NK; ++n) s += san2[n];
    sc = 1.0f / s;
  }
  __syncthreads();
  if (tid < NK) san2[tid] *= sc;
  __syncthreads();
  {  // c_g2
    float acc = 0.f;
    for (int n = 0; n < NK; ++n)
      acc += san2[n] * encn[((size_t)b * NK + n) * H + tid];
    cg2_out[b * H + tid] = acc;
  }
  // a_n2t = an @ n2t, renormalized
  for (int t = tid; t < TK; t += blockDim.x) {
    float acc = 0.f;
    for (int n = 0; n < NK; ++n)
      acc += san[n] * n2t[((size_t)b * NK + n) * TK + t];
    sat[t] = acc;
  }
  __syncthreads();
  if (tid == 0) {
    float s = 0.f;
    for (int t = 0; t < TK; ++t) s += sat[t];
    sc = 1.0f / s;
  }
  __syncthreads();
  for (int t = tid; t < TK; t += blockDim.x)
    an2t_out[b * TK + t] = sat[t] * sc;
  // flow_next: per-node max over tokens of a*n2t, then normalize
  if (tid < NK) {
    float mx = -1e30f;
    for (int t = 0; t < TK; ++t)
      mx = fmaxf(mx, a_doc[b * TK + t] * n2t[((size_t)b * NK + tid) * TK + t]);
    san2[tid] = mx;
  }
  __syncthreads();
  if (tid == 0) {
    float s = 0.f;
    for (int n = 0; n < NK; ++n) s += san2[n];
    sc = 1.0f / s;
  }
  __syncthreads();
  if (tid < NK) flow_out[b * NK + tid] = san2[tid] * sc;
}

__global__ void k_pgen(const float* __restrict__ cd, const float* __restrict__ cg,
                       const float* __restrict__ cg2, const float* __restrict__ s_hat,
                       const float* __restrict__ x, const float* __restrict__ Wpg,
                       const float* __restrict__ bpg, float* __restrict__ pgen) {
  __shared__ float red[256];
  int b = blockIdx.x, tid = threadIdx.x;
  float acc = 0.f;
#pragma unroll
  for (int j = 0; j < 6; ++j) {  // 5H+E = 1536 = 6*256
    int idx = tid + j * 256;
    float xv;
    if (idx < H)            xv = cd[b * H + idx];
    else if (idx < 2 * H)   xv = cg[b * H + idx - H];
    else if (idx < 3 * H)   xv = cg2[b * H + idx - 2 * H];
    else if (idx < 5 * H)   xv = s_hat[b * 2 * H + idx - 3 * H];
    else                    xv = x[b * E + idx - 5 * H];
    acc += xv * Wpg[idx];
  }
  red[tid] = acc;
  __syncthreads();
  for (int s = 128; s > 0; s >>= 1) {
    if (tid < s) red[tid] += red[tid + s];
    __syncthreads();
  }
  if (tid == 0) pgen[b] = sigmoidf(red[0] + bpg[0]);
}

__global__ void k_build_hcat(const float* __restrict__ hn, const float* __restrict__ cd,
                             const float* __restrict__ cg, const float* __restrict__ cg2,
                             float* __restrict__ hcat) {
  int b = blockIdx.x, h = threadIdx.x;
  float* row = hcat + (size_t)b * 4 * H;
  row[h]         = hn[b * H + h];
  row[H + h]     = cd[b * H + h];
  row[2 * H + h] = cg[b * H + h];
  row[3 * H + h] = cg2[b * H + h];
}

// In-place: row holds logits; replace with p_gen * softmax(row).
__global__ void k_vocab_softmax(float* __restrict__ fd, const float* __restrict__ pgen) {
  __shared__ float red[512];
  __shared__ float sc;
  int b = blockIdx.x, tid = threadIdx.x;
  float* row = fd + (size_t)b * V;
  float m = -1e30f;
  for (int i = tid; i < V; i += 512) m = fmaxf(m, row[i]);
  red[tid] = m;
  __syncthreads();
  for (int s = 256; s > 0; s >>= 1) {
    if (tid < s) red[tid] = fmaxf(red[tid], red[tid + s]);
    __syncthreads();
  }
  if (tid == 0) sc = red[0];
  __syncthreads();
  m = sc;
  float sum = 0.f;
  for (int i = tid; i < V; i += 512) sum += expf(row[i] - m);
  __syncthreads();
  red[tid] = sum;
  __syncthreads();
  for (int s = 256; s > 0; s >>= 1) {
    if (tid < s) red[tid] += red[tid + s];
    __syncthreads();
  }
  if (tid == 0) sc = pgen[b] / red[0];
  __syncthreads();
  float scale = sc;
  for (int i = tid; i < V; i += 512) row[i] = expf(row[i] - m) * scale;
}

// Serial per-row scatter-add (no float atomics -> bitwise deterministic).
__global__ void k_scatter(const float* __restrict__ a, const int* __restrict__ voc,
                          const float* __restrict__ pgen, float* __restrict__ fd) {
  int b = blockIdx.x;
  if (threadIdx.x != 0) return;
  float w = 1.0f - pgen[b];
  float* row = fd + (size_t)b * V;
  for (int t = 0; t < TK; ++t) row[voc[b * TK + t]] += w * a[b * TK + t];
}

// ---------------------------------------------------------------------------
extern "C" void kernel_launch(void* const* d_in, const int* in_sizes, int n_in,
                              void* d_out_v, int out_size, void* d_ws, size_t ws_size,
                              hipStream_t stream) {
  (void)in_sizes; (void)n_in; (void)out_size; (void)ws_size;
  const int*   y_t_1  = (const int*)d_in[0];
  const float* h_prev = (const float*)d_in[1];
  const float* c_prev = (const float*)d_in[2];
  const float* enc    = (const float*)d_in[3];
  const float* enc_n  = (const float*)d_in[4];
  const float* mask_t = (const float*)d_in[5];
  const float* mask_n = (const float*)d_in[6];
  const float* c_t_d  = (const float*)d_in[7];
  const float* c_t_g  = (const float*)d_in[8];
  const float* c_t_g2 = (const float*)d_in[9];
  const float* cover  = (const float*)d_in[10];
  const float* flow   = (const float*)d_in[11];
  const float* n2t    = (const float*)d_in[12];
  const float* graph  = (const float*)d_in[13];
  const int*   voc_d  = (const int*)d_in[14];
  const float* emb    = (const float*)d_in[16];
  const float* Wx  = (const float*)d_in[17];
  const float* bx  = (const float*)d_in[18];
  const float* Wi  = (const float*)d_in[19];
  const float* Wh  = (const float*)d_in[20];
  const float* bi  = (const float*)d_in[21];
  const float* bh  = (const float*)d_in[22];
  const float* Whd = (const float*)d_in[23];
  const float* Wsd = (const float*)d_in[24];
  const float* wcd = (const float*)d_in[25];
  const float* vd  = (const float*)d_in[26];
  const float* bd  = (const float*)d_in[27];
  const float* Wn  = (const float*)d_in[28];
  const float* Wsf = (const float*)d_in[29];
  const float* vf  = (const float*)d_in[30];
  const float* bfv = (const float*)d_in[31];
  const float* Wpg = (const float*)d_in[32];
  const float* bpg = (const float*)d_in[33];
  const float* W1  = (const float*)d_in[34];
  const float* b1  = (const float*)d_in[35];
  const float* W2  = (const float*)d_in[36];
  const float* b2  = (const float*)d_in[37];

  // Output layout (floats, reference return order)
  float* out    = (float*)d_out_v;
  float* fd     = out;                        // [B,V]
  float* h_new  = fd + (size_t)B_ * V;        // [B,H]
  float* c_new  = h_new + B_ * H;             // [B,H]
  float* cd_o   = c_new + B_ * H;             // [B,H]
  float* cg_o   = cd_o + B_ * H;              // [B,H]
  float* cg2_o  = cg_o + B_ * H;              // [B,H]
  float* a_o    = cg2_o + B_ * H;             // [B,TK]
  float* an2t_o = a_o + B_ * TK;              // [B,TK]
  float* pgen_o = an2t_o + B_ * TK;           // [B,1]
  float* cov_o  = pgen_o + B_;                // [B,TK]
  float* flow_o = cov_o + B_ * TK;            // [B,NK]

  // Workspace layout (~2.6 MB total)
  float* ws    = (float*)d_ws;
  float* xcat  = ws;                          // [B,4H]
  float* x     = xcat + B_ * 4 * H;           // [B,E]
  float* gates = x + B_ * E;                  // [B,4H]
  float* s_hat = gates + B_ * 4 * H;          // [B,2H]
  float* sWsd  = s_hat + B_ * 2 * H;          // [B,H]
  float* sWsf  = sWsd + B_ * H;               // [B,H]
  float* e_doc = sWsf + B_ * H;               // [B,TK]
  float* e_nod = e_doc + B_ * TK;             // [B,NK]
  float* hcat  = e_nod + B_ * NK;             // [B,4H]
  float* out1  = hcat + B_ * 4 * H;           // [B,H]

  auto gemm = [&](const float* A, int lda, const float* Bm, int ldb,
                  const float* A2, int lda2, const float* B2, int ldb2,
                  const float* bias, const float* bias2,
                  float* C, int ldc, int M, int N, int K, int K2) {
    int tiles = (M / 16) * (N / 16);
    int blocks = (tiles + 7) / 8;
    k_wmma_gemm<<<blocks, 256, 0, stream>>>(A, lda, Bm, ldb, A2, lda2, B2, ldb2,
                                            bias, bias2, C, ldc, M, N, K, K2);
  };

  // 1. xcat = [c_t_d, c_t_g, c_t_g2, emb[y]]
  k_build_xcat<<<B_, H, 0, stream>>>(c_t_d, c_t_g, c_t_g2, y_t_1, emb, xcat);
  // 2. x = xcat @ Wx + bx
  gemm(xcat, 4 * H, Wx, E, nullptr, 0, nullptr, 0, bx, nullptr,
       x, E, B_, E, 4 * H, 0);
  // 3. gates = x @ Wi + h_prev @ Wh + bi + bh
  gemm(x, E, Wi, 4 * H, h_prev, H, Wh, 4 * H, bi, bh,
       gates, 4 * H, B_, 4 * H, E, H);
  // 4. LSTM cell -> h_new, c_new (out) + s_hat (ws)
  k_lstm<<<B_, H, 0, stream>>>(gates, c_prev, h_new, c_new, s_hat);
  // 5. sWsd = s_hat @ Wsd ; sWsf = s_hat @ Wsf
  gemm(s_hat, 2 * H, Wsd, H, nullptr, 0, nullptr, 0, nullptr, nullptr,
       sWsd, H, B_, H, 2 * H, 0);
  gemm(s_hat, 2 * H, Wsf, H, nullptr, 0, nullptr, 0, nullptr, nullptr,
       sWsf, H, B_, H, 2 * H, 0);
  // 6. Fused attention scores (doc: 3200 tiles, node: 400 tiles)
  {
    int tiles = (B_ * TK) / 16;
    k_attn_score<<<(tiles + 7) / 8, 256, 0, stream>>>(
        enc, Whd, sWsd, cover, wcd, bd, vd, e_doc, B_ * TK, TK);
  }
  {
    int tiles = (B_ * NK) / 16;
    k_attn_score<<<(tiles + 7) / 8, 256, 0, stream>>>(
        enc_n, Wn, sWsf, nullptr, nullptr, bfv, vf, e_nod, B_ * NK, NK);
  }
  // 7. Doc softmax + coverage + c_d
  k_doc_attn<<<B_, 256, 0, stream>>>(e_doc, mask_t, cover, enc, a_o, cov_o, cd_o);
  // 8. Node pipeline: an, c_g, an2, c_g2, a_n2t, flow_next
  k_node_attn<<<B_, 256, 0, stream>>>(e_nod, mask_n, flow, enc_n, graph, n2t,
                                      a_o, cg_o, cg2_o, an2t_o, flow_o);
  // 9. p_gen
  k_pgen<<<B_, 256, 0, stream>>>(cd_o, cg_o, cg2_o, s_hat, x, Wpg, bpg, pgen_o);
  // 10. out = [h_new, c_d, c_g, c_g2] @ W1 + b1
  k_build_hcat<<<B_, H, 0, stream>>>(h_new, cd_o, cg_o, cg2_o, hcat);
  gemm(hcat, 4 * H, W1, H, nullptr, 0, nullptr, 0, b1, nullptr,
       out1, H, B_, H, 4 * H, 0);
  // 11. logits = out @ W2 + b2, written straight into final_dist
  gemm(out1, H, W2, V, nullptr, 0, nullptr, 0, b2, nullptr,
       fd, V, B_, V, H, 0);
  // 12. final_dist = p_gen * softmax(logits)  (in place)
  k_vocab_softmax<<<B_, 512, 0, stream>>>(fd, pgen_o);
  // 13. final_dist[b, voc_d[b,t]] += (1-p_gen) * a[b,t]  (deterministic)
  k_scatter<<<B_, 32, 0, stream>>>(a_o, voc_d, pgen_o, fd);
}
